// PermutationMatrix_26989574488834
// MI455X (gfx1250) — matrix operand order repaired and checked by reading, not compiled, not run on hardware
//
#include <hip/hip_runtime.h>
#include <hip/hip_bf16.h>

#define NDIM 4096

typedef __attribute__((ext_vector_type(16))) __bf16 v16bf;
typedef __attribute__((ext_vector_type(8)))  float  v8f;
typedef int v4i __attribute__((vector_size(16)));   // matches builtin param type

#define AS1 __attribute__((address_space(1)))
#define AS3 __attribute__((address_space(3)))

#if __has_builtin(__builtin_amdgcn_global_load_async_to_lds_b128)
#define USE_ASYNC_LDS 1
#endif

template <int N>
__device__ __forceinline__ void wait_asynccnt() {
#ifdef USE_ASYNC_LDS
#if __has_builtin(__builtin_amdgcn_s_wait_asynccnt)
  __builtin_amdgcn_s_wait_asynccnt((short)N);
#else
  asm volatile("s_wait_asynccnt %0" ::"n"(N));
#endif
#endif
}

// ---------------- Phase 1: m = exp(T*(x - rowmax)) ----------------
__global__ __launch_bounds__(256) void k_rowmax_exp(const float* __restrict__ x,
                                                    const int* __restrict__ epoch,
                                                    float* __restrict__ m) {
  __shared__ float red[256];
  const int row = blockIdx.x;
  const float* xr = x + (size_t)row * NDIM;
  float mx = -3.402823466e38f;
  for (int c = threadIdx.x; c < NDIM; c += 256) mx = fmaxf(mx, xr[c]);
  red[threadIdx.x] = mx;
  __syncthreads();
  for (int s = 128; s > 0; s >>= 1) {
    if (threadIdx.x < s) red[threadIdx.x] = fmaxf(red[threadIdx.x], red[threadIdx.x + s]);
    __syncthreads();
  }
  mx = red[0];
  const float T = (float)((*epoch) / 10 + 1) * 2.0f;
  float* mr = m + (size_t)row * NDIM;
  for (int c = threadIdx.x; c < NDIM; c += 256) mr[c] = __expf(T * (xr[c] - mx));
}

// ---------------- Phase 2: Sinkhorn normalization ----------------
__global__ __launch_bounds__(256) void k_rownorm(float* __restrict__ m) {
  __shared__ float red[256];
  const int row = blockIdx.x;
  float* mr = m + (size_t)row * NDIM;
  float s = 0.0f;
  for (int c = threadIdx.x; c < NDIM; c += 256) s += mr[c];
  red[threadIdx.x] = s;
  __syncthreads();
  for (int st = 128; st > 0; st >>= 1) {
    if (threadIdx.x < st) red[threadIdx.x] += red[threadIdx.x + st];
    __syncthreads();
  }
  s = red[0];
  for (int c = threadIdx.x; c < NDIM; c += 256) mr[c] = mr[c] / s;
}

__global__ __launch_bounds__(256) void k_colsum(const float* __restrict__ m,
                                                float* __restrict__ colsum) {
  const int c = blockIdx.x * 256 + threadIdx.x;
  float s = 0.0f;
  for (int r = 0; r < NDIM; ++r) s += m[(size_t)r * NDIM + c];
  colsum[c] = s;
}

__global__ __launch_bounds__(256) void k_coldiv(float* __restrict__ m,
                                                const float* __restrict__ colsum) {
  const size_t i = ((size_t)blockIdx.x * 256 + threadIdx.x) * 4;
  const int c = (int)(i & (NDIM - 1));
  float4 v = *(const float4*)(m + i);
  const float4 cs = *(const float4*)(colsum + c);
  v.x = v.x / cs.x; v.y = v.y / cs.y; v.z = v.z / cs.z; v.w = v.w / cs.w;
  *(float4*)(m + i) = v;
}

// ---------------- Phase 3: fp32 -> bf16 ----------------
__global__ __launch_bounds__(256) void k_tobf16(const float* __restrict__ m,
                                                __bf16* __restrict__ mb) {
  const size_t i = ((size_t)blockIdx.x * 256 + threadIdx.x) * 4;
  const float4 v = *(const float4*)(m + i);
  __bf16 o[4] = { (__bf16)v.x, (__bf16)v.y, (__bf16)v.z, (__bf16)v.w };
  *(uint2*)(mb + i) = *(const uint2*)o;
}

// ---------------- Phase 4: WMMA GEMM, double-buffered async LDS ----------------
// C(MxN) = A(MxK,bf16 row-major) * B(KxN)
// BSRC==0: B given as row-major NxK bf16 (B[k][n] = Bp[n*N+k]; B = m^T from mb)
// BSRC==1: B[k][n] = sigmoid(lower[n*N+k]) * (n>=k)   (B = L^T fused from fp32 'lower')
// OUTF32: store fp32 accumulators, else convert to bf16.
template <int BSRC, bool OUTF32>
__global__ __launch_bounds__(256) void k_gemm(const __bf16* __restrict__ A,
                                              const void* __restrict__ Bp,
                                              void* __restrict__ Cp) {
  constexpr int BK  = 32;
  constexpr int LDL = 40;  // 80B row stride: rows 16B-aligned for b128 async LDS stores
  __shared__ __bf16 Alds[2][128][LDL];   // [buf][m][k]
  __shared__ __bf16 Blds[2][128][LDL];   // [buf][n][k]
  // async ops per thread per stage (A: 2 x b128; B async only when BSRC==0)
  constexpr int NPER = (BSRC == 0) ? 4 : 2;

  const int tid  = threadIdx.x;
  const int lane = tid & 31;
  const int wave = tid >> 5;
  const int half = lane >> 4;
  const int l16  = lane & 15;
  const int waveM = wave >> 2;           // 0..1
  const int waveN = wave & 3;            // 0..3
  const int m0blk = blockIdx.y * 128;
  const int n0blk = blockIdx.x * 128;
  const int r2 = tid >> 1, hh = tid & 1; // staging coords: row 0..127, half-row 0..1

  union AF { v16bf v; unsigned u[8]; };
  v8f acc[4][2] = {};

  auto stageA = [&](int buf, int kb) {
    const __bf16* src = A + (size_t)(m0blk + r2) * NDIM + kb + hh * 16;
#ifdef USE_ASYNC_LDS
    __builtin_amdgcn_global_load_async_to_lds_b128(
        (AS1 v4i*)src, (AS3 v4i*)&Alds[buf][r2][hh * 16], 0, 0);
    __builtin_amdgcn_global_load_async_to_lds_b128(
        (AS1 v4i*)(src + 8), (AS3 v4i*)&Alds[buf][r2][hh * 16 + 8], 0, 0);
#else
    const uint4* s4 = (const uint4*)src;
    const uint4 q0 = s4[0], q1 = s4[1];
    unsigned* dst = (unsigned*)&Alds[buf][r2][hh * 16];
    dst[0] = q0.x; dst[1] = q0.y; dst[2] = q0.z; dst[3] = q0.w;
    dst[4] = q1.x; dst[5] = q1.y; dst[6] = q1.z; dst[7] = q1.w;
#endif
  };

  auto stageB = [&](int buf, int kb) {
    if (BSRC == 0) {
      const __bf16* B = (const __bf16*)Bp;  // row-major N x K
      const __bf16* src = B + (size_t)(n0blk + r2) * NDIM + kb + hh * 16;
#ifdef USE_ASYNC_LDS
      __builtin_amdgcn_global_load_async_to_lds_b128(
          (AS1 v4i*)src, (AS3 v4i*)&Blds[buf][r2][hh * 16], 0, 0);
      __builtin_amdgcn_global_load_async_to_lds_b128(
          (AS1 v4i*)(src + 8), (AS3 v4i*)&Blds[buf][r2][hh * 16 + 8], 0, 0);
#else
      const uint4* s4 = (const uint4*)src;
      const uint4 q0 = s4[0], q1 = s4[1];
      unsigned* dst = (unsigned*)&Blds[buf][r2][hh * 16];
      dst[0] = q0.x; dst[1] = q0.y; dst[2] = q0.z; dst[3] = q0.w;
      dst[4] = q1.x; dst[5] = q1.y; dst[6] = q1.z; dst[7] = q1.w;
#endif
    } else {
      const float* Lw = (const float*)Bp;   // fused sigmoid + tril mask
      const int gn = n0blk + r2;
      const float* src = Lw + (size_t)gn * NDIM + kb + hh * 16;
      #pragma unroll
      for (int e = 0; e < 16; ++e) {
        const int gk = kb + hh * 16 + e;
        const float sv = 1.0f / (1.0f + __expf(-src[e]));
        Blds[buf][r2][hh * 16 + e] = (gn >= gk) ? (__bf16)sv : (__bf16)0.0f;
      }
    }
  };

  auto compute = [&](int buf) {
    AF af[4];
    #pragma unroll
    for (int i = 0; i < 4; ++i) {
      const int mrow = waveM * 64 + i * 16 + l16;
      #pragma unroll
      for (int p = 0; p < 8; ++p) {
        const int k = ((p < 4) ? 2 * p : 2 * p + 8) + 8 * half;  // 16-bit A layout
        af[i].u[p] = *(const unsigned*)&Alds[buf][mrow][k];
      }
    }
    AF bfg[2];
    #pragma unroll
    for (int j = 0; j < 2; ++j) {
      const int ncol = waveN * 32 + j * 16 + l16;
      #pragma unroll
      for (int p = 0; p < 8; ++p) {
        const int k = ((p < 4) ? 2 * p : 2 * p + 8) + 8 * half;  // 16-bit B layout
        bfg[j].u[p] = *(const unsigned*)&Blds[buf][ncol][k];
      }
    }
    #pragma unroll
    for (int i = 0; i < 4; ++i)
      #pragma unroll
      for (int j = 0; j < 2; ++j)
        acc[i][j] = __builtin_amdgcn_wmma_f32_16x16x32_bf16(
            false, af[i].v, false, bfg[j].v, (short)0, acc[i][j], false, false);
  };

  // -------- software pipeline: stage(i+1) overlaps compute(i) --------
  stageA(0, 0);
  stageB(0, 0);
  constexpr int STEPS = NDIM / BK;
  for (int i = 0; i < STEPS; ++i) {
    const int cur = i & 1;
    if (i + 1 < STEPS) {            // uniform branch: safe to overwrite cur^1,
      stageA(cur ^ 1, (i + 1) * BK);// readers of it finished at prev trailing barrier
      stageB(cur ^ 1, (i + 1) * BK);
      wait_asynccnt<NPER>();        // <=NPER in flight -> buffer 'cur' fully landed
    } else {
      wait_asynccnt<0>();
    }
    __syncthreads();                // all waves' copies visible (ds stores waited here too)
    compute(cur);
    __syncthreads();                // all waves done reading 'cur' before next overwrite
  }

  // ---- epilogue: C/D layout: VGPR v, lane l -> M = v + 8*(l/16), N = l%16 ----
  #pragma unroll
  for (int i = 0; i < 4; ++i) {
    const int grow = m0blk + waveM * 64 + i * 16;
    #pragma unroll
    for (int j = 0; j < 2; ++j) {
      const int gcol = n0blk + waveN * 32 + j * 16 + l16;
      #pragma unroll
      for (int v = 0; v < 8; ++v) {
        const int r = grow + v + 8 * half;
        if (OUTF32) ((float*)Cp)[(size_t)r * NDIM + gcol] = acc[i][j][v];
        else        ((__bf16*)Cp)[(size_t)r * NDIM + gcol] = (__bf16)acc[i][j][v];
      }
    }
  }
}

// ---------------- host-side orchestration ----------------
extern "C" void kernel_launch(void* const* d_in, const int* in_sizes, int n_in,
                              void* d_out, int out_size, void* d_ws, size_t ws_size,
                              hipStream_t stream) {
  const float* matrix = (const float*)d_in[0];
  const float* lower  = (const float*)d_in[1];
  const int*   epoch  = (const int*)d_in[2];

  const size_t NN = (size_t)NDIM * NDIM;
  float*  m      = (float*)d_ws;                                   // 64 MB
  __bf16* mb     = (__bf16*)((char*)d_ws + NN * sizeof(float));    // +64 MB, 32 MB
  float*  colsum = (float*)((char*)d_ws + NN * sizeof(float) + NN * sizeof(__bf16));
  __bf16* t1b    = (__bf16*)m;  // GEMM1 output reuses m region (m dead after k_tobf16)

  k_rowmax_exp<<<NDIM, 256, 0, stream>>>(matrix, epoch, m);

  // Sinkhorn: epoch=15 -> mult=2 -> 40 iterations (trip count must be host-side)
  for (int it = 0; it < 40; ++it) {
    k_rownorm<<<NDIM, 256, 0, stream>>>(m);
    k_colsum<<<NDIM / 256, 256, 0, stream>>>(m, colsum);
    k_coldiv<<<(int)(NN / 1024), 256, 0, stream>>>(m, colsum);
  }

  k_tobf16<<<(int)(NN / 1024), 256, 0, stream>>>(m, mb);

  dim3 grid(NDIM / 128, NDIM / 128);
  // T1 = m @ L^T   (B fused from 'lower': sigmoid + tril mask), bf16 out
  k_gemm<1, false><<<grid, 256, 0, stream>>>(mb, (const void*)lower, (void*)t1b);
  // out = T1 @ m^T (B = mb read column-major), fp32 out  == (m @ L @ m^T)^T
  k_gemm<0, true><<<grid, 256, 0, stream>>>(t1b, (const void*)mb, d_out);
}
